// Translator_37890201485660
// MI455X (gfx1250) — matrix operand order repaired and checked by reading, not compiled, not run on hardware
//
#include <hip/hip_runtime.h>
#include <hip/hip_bf16.h>

#define BATCH  32
#define TX     64
#define TY     32
#define EMB    256
#define UNITS  512
#define DMODEL 1024   // 2*UNITS
#define ATT    512
#define VTAR   32000
#define BK     128    // K-chunk staged in LDS by the TDM GEMM

typedef __bf16 bf16_t;
typedef __attribute__((ext_vector_type(16))) __bf16 bf16x16;
typedef __attribute__((ext_vector_type(8)))  __bf16 bf16x8;
typedef __attribute__((ext_vector_type(8)))  float  floatx8;
typedef __attribute__((ext_vector_type(4)))  unsigned int uint32x4;
typedef __attribute__((ext_vector_type(8)))  int   int32x8;
typedef __attribute__((ext_vector_type(4)))  int   int32x4;

#if __has_builtin(__builtin_amdgcn_tensor_load_to_lds)
#define HAVE_TDM 1
#if __has_include(<hip/amd_detail/amd_gfx1250_TDM.h>)
#define TDM_6ARG 1   // amdgpu-toolchain (clang-23 + therock headers): 6-arg builtin
#endif
#endif

// The TDM builtin only consumes descriptor integers, so clang would otherwise
// prove the LDS tile is never written and fold its loads to undef. Escaping
// the LDS pointer into an asm with a memory clobber makes the asm a potential
// writer of that object, so the loads must materialize (as plain ds ops —
// volatile is NOT used, since it forced serialized SCOPE_SYS flat loads).
#define LDS_PRODUCED(p) asm volatile("" : : "r"((const void*)(p)) : "memory")

__device__ __forceinline__ float sigmoidf_(float x) { return 1.0f / (1.0f + expf(-x)); }

#ifdef HAVE_TDM
// Build a 2D Tensor-DMA descriptor (CDNA5 ISA §8.3/8.4) and issue TENSOR_LOAD_TO_LDS.
__device__ __forceinline__ void tdm_load_2d(unsigned lds_off, const void* gptr,
                                            unsigned tile_w, unsigned tile_h,
                                            unsigned row_stride_elems)
{
    unsigned long long ga = (unsigned long long)gptr;
    uint32x4 g0;
    g0.x = 1u;                                             // count=1, user descriptor
    g0.y = lds_off;                                        // lds_addr (bytes)
    g0.z = (unsigned)(ga & 0xffffffffu);                   // global_addr[31:0]
    g0.w = (unsigned)((ga >> 32) & 0x01ffffffu) | (2u << 30); // global_addr[56:32] | type=2

    const unsigned td0 = row_stride_elems;                 // tensor_dim0 (elements)
    const unsigned td1 = 0x7fffffffu;                      // tensor_dim1: huge -> no OOB clip
    int32x8 g1;
    g1[0] = (int)(1u << 16);                               // workgroup_mask=0, data_size=1 (2B)
    g1[1] = (int)((td0 & 0xffffu) << 16);                  // tensor_dim0[15:0]
    g1[2] = (int)((td0 >> 16) | ((td1 & 0xffffu) << 16));  // dim0[31:16] | dim1[15:0]
    g1[3] = (int)((td1 >> 16) | (tile_w << 16));           // dim1[31:16] | tile_dim0
    g1[4] = (int)tile_h;                                   // tile_dim1 | tile_dim2=0
    g1[5] = (int)row_stride_elems;                         // tensor_dim0_stride[31:0]
    g1[6] = 0;                                             // stride0[47:32] | stride1[15:0]
    g1[7] = 0;

    int32x4 z4 = {0, 0, 0, 0};
#ifdef TDM_6ARG
    int32x8 z8 = {0, 0, 0, 0, 0, 0, 0, 0};
    __builtin_amdgcn_tensor_load_to_lds(g0, g1, z4, z4, z8, 0);
#else
    __builtin_amdgcn_tensor_load_to_lds(g0, g1, z4, z4, 0);
#endif
}
#endif

// ---------------------------------------------------------------------------
// Direct-global WMMA GEMM (latency path for the small recurrent GEMMs).
// LDB is a compile-time constant so the 16 strided B-fragment loads become
// base + immediate-offset loads packed into one s_clause.
// ---------------------------------------------------------------------------
template <int LDB>
__global__ void __launch_bounds__(256)
gemm_bf16_wmma(const bf16_t* __restrict__ A, int lda,
               const bf16_t* __restrict__ B,
               float* __restrict__ C, int ldc,
               const float* __restrict__ bias, int K)
{
    const int lane = threadIdx.x & 31;
    const int wave = threadIdx.x >> 5;
    const int row0 = blockIdx.y * 32 + (wave >> 2) * 16;
    const int col0 = blockIdx.x * 64 + (wave & 3) * 16;
    const int half = lane >> 4;
    const int lid  = lane & 15;

    const bf16_t* Ap = A + (long)(row0 + lid) * lda + half * 8;
    const bf16_t* Bp = B + (long)(half * 8) * LDB + (col0 + lid);

    floatx8 acc = {};
    for (int k0 = 0; k0 < K; k0 += 32) {
        bf16x16 a, b;
#pragma unroll
        for (int e = 0; e < 8; ++e) { a[e] = Ap[e]; a[e + 8] = Ap[e + 16]; }
#pragma unroll
        for (int e = 0; e < 8; ++e) { b[e] = Bp[e * LDB]; b[e + 8] = Bp[(e + 16) * LDB]; }   // imm offsets
        __builtin_prefetch(Bp + 32 * LDB, 0, 1);
        acc = __builtin_amdgcn_wmma_f32_16x16x32_bf16(false, a, false, b,
                                                      (short)0, acc, false, false);
        Ap += 32;
        Bp += 32 * LDB;
    }

    const float bv = bias ? bias[col0 + lid] : 0.0f;
    float* Cp = C + (long)(row0 + half * 8) * ldc + (col0 + lid);
#pragma unroll
    for (int e = 0; e < 8; ++e) Cp[(long)e * ldc] = acc[e] + bv;
}

// ---------------------------------------------------------------------------
// TDM-staged WMMA GEMM (bandwidth path for batched/streaming GEMMs):
// per 128-deep K-chunk, wave0 issues two tensor_load_to_lds DMAs (A 32xBK,
// B BKx64), s_wait_tensorcnt, barrier, then 8 waves run WMMA from LDS.
// ---------------------------------------------------------------------------
__global__ void __launch_bounds__(256)
gemm_bf16_tdm(const bf16_t* __restrict__ A, int lda,
              const bf16_t* __restrict__ B, int ldb,
              float* __restrict__ C, int ldc,
              const float* __restrict__ bias, int K)
{
    __shared__ __align__(16) char smem[(32 * BK + BK * 64) * 2];
    bf16_t* sA = (bf16_t*)smem;                       // [32][BK]
    bf16_t* sB = (bf16_t*)(smem + 32 * BK * 2);       // [BK][64]
    const unsigned OFF_A = 0;
    const unsigned OFF_B = 32 * BK * 2;

    const int lane = threadIdx.x & 31;
    const int wave = threadIdx.x >> 5;
    const int half = lane >> 4;
    const int lid  = lane & 15;
    const int lr   = (wave >> 2) * 16 + lid;          // local row in 0..31
    const int lc   = (wave & 3) * 16 + lid;           // local col in 0..63
    const long arow0 = (long)blockIdx.y * 32;
    const long bcol0 = (long)blockIdx.x * 64;

    floatx8 acc = {};

    for (int kc = 0; kc < K; kc += BK) {
        LDS_PRODUCED(smem);                           // order consumers before reuse
        __syncthreads();                              // protect LDS from previous chunk
#ifdef HAVE_TDM
        if (threadIdx.x < 32) {
            tdm_load_2d(OFF_A, A + arow0 * lda + kc, BK, 32, (unsigned)lda);
            tdm_load_2d(OFF_B, B + (long)kc * ldb + bcol0, 64, BK, (unsigned)ldb);
            __builtin_amdgcn_s_wait_tensorcnt(0);
        }
#else
        for (int i = threadIdx.x; i < (32 * BK) / 8; i += 256) {
            int m = (i * 8) / BK, k = (i * 8) % BK;
            *(bf16x8*)(sA + m * BK + k) =
                *(const bf16x8*)(A + (arow0 + m) * lda + kc + k);
        }
        for (int i = threadIdx.x; i < (BK * 64) / 8; i += 256) {
            int k = (i * 8) / 64, n = (i * 8) % 64;
            *(bf16x8*)(sB + k * 64 + n) =
                *(const bf16x8*)(B + (long)(kc + k) * ldb + bcol0 + n);
        }
#endif
        LDS_PRODUCED(smem);                           // TDM wrote LDS: loads must reload
        __syncthreads();

#pragma unroll
        for (int ks = 0; ks < BK; ks += 32) {
            bf16x16 a, b;
            const bf16_t* ap = sA + lr * BK + ks + half * 8;
            bf16x8 a0 = *(const bf16x8*)(ap);
            bf16x8 a1 = *(const bf16x8*)(ap + 16);
#pragma unroll
            for (int e = 0; e < 8; ++e) { a[e] = a0[e]; a[e + 8] = a1[e]; }
#pragma unroll
            for (int e = 0; e < 16; ++e) b[e] = sB[(ks + half * 8 + (e & 7) + (e >> 3) * 16) * 64 + lc];
            acc = __builtin_amdgcn_wmma_f32_16x16x32_bf16(false, a, false, b,
                                                          (short)0, acc, false, false);
        }
    }

    const int col0 = blockIdx.x * 64 + (wave & 3) * 16;
    const int row0 = blockIdx.y * 32 + (wave >> 2) * 16;
    const float bv = bias ? bias[col0 + lid] : 0.0f;
    float* Cp = C + (long)(row0 + half * 8) * ldc + (col0 + lid);
#pragma unroll
    for (int e = 0; e < 8; ++e) Cp[(long)e * ldc] = acc[e] + bv;
}

// ---------------------------------------------------------------------------
// Elementwise / utility kernels
// ---------------------------------------------------------------------------
__global__ void cvt_f32_bf16(const float* __restrict__ s, bf16_t* __restrict__ d, long n) {
    long i = blockIdx.x * 256L + threadIdx.x;
    long stride = (long)gridDim.x * 256L;
    for (; i < n; i += stride) d[i] = (bf16_t)s[i];
}

__global__ void zero_f32(float* p, long n) {
    long i = blockIdx.x * 256L + threadIdx.x;
    long stride = (long)gridDim.x * 256L;
    for (; i < n; i += stride) p[i] = 0.0f;
}

__global__ void zero_bf16(bf16_t* p, long n) {
    long i = blockIdx.x * 256L + threadIdx.x;
    long stride = (long)gridDim.x * 256L;
    for (; i < n; i += stride) p[i] = (bf16_t)0.0f;
}

__global__ void gather_embed(const float* __restrict__ emb, const int* __restrict__ toks,
                             bf16_t* __restrict__ out, int ntok, int E) {
    long idx = blockIdx.x * 256L + threadIdx.x;
    if (idx >= (long)ntok * E) return;
    int t = (int)(idx / E), e = (int)(idx % E);
    out[idx] = (bf16_t)emb[(long)toks[t] * E + e];
}

// Fused LSTM gate update (Keras i,f,g,o order): z = xwx_t + hwh + b
__global__ void __launch_bounds__(256)
lstm_gates(const float* __restrict__ xwx_t, int xs,
           const float* __restrict__ hwh,
           const float* __restrict__ bias,
           float* __restrict__ h, bf16_t* __restrict__ hbf, float* __restrict__ c,
           float* __restrict__ y_t, long ys, int U, int n)
{
    int idx = blockIdx.x * 256 + threadIdx.x;
    if (idx >= n) return;
    int b = idx / U, u = idx - b * U;
    const float* xr = xwx_t + (long)b * xs;
    const float* hr = hwh + (long)b * 4 * U;
    float zi = xr[u]         + hr[u]         + bias[u];
    float zf = xr[U + u]     + hr[U + u]     + bias[U + u];
    float zg = xr[2 * U + u] + hr[2 * U + u] + bias[2 * U + u];
    float zo = xr[3 * U + u] + hr[3 * U + u] + bias[3 * U + u];
    float cn = sigmoidf_(zf) * c[idx] + sigmoidf_(zi) * tanhf(zg);
    float hn = sigmoidf_(zo) * tanhf(cn);
    c[idx] = cn;
    h[idx] = hn;
    hbf[idx] = (bf16_t)hn;
    if (y_t) y_t[(long)b * ys + u] = hn;
}

__global__ void concat_hc(const float* __restrict__ hf, const float* __restrict__ cf,
                          const float* __restrict__ hb, const float* __restrict__ cb,
                          float* __restrict__ h, bf16_t* __restrict__ hbf, float* __restrict__ c) {
    int idx = blockIdx.x * 256 + threadIdx.x;
    if (idx >= BATCH * DMODEL) return;
    int b = idx / DMODEL, j = idx - b * DMODEL;
    float hv = (j < UNITS) ? hf[b * UNITS + j] : hb[b * UNITS + j - UNITS];
    float cv = (j < UNITS) ? cf[b * UNITS + j] : cb[b * UNITS + j - UNITS];
    h[idx] = hv; hbf[idx] = (bf16_t)hv; c[idx] = cv;
}

__global__ void init_tok(const int* __restrict__ targ, int* __restrict__ tok) {
    int b = threadIdx.x;
    if (b < BATCH) tok[b] = targ[b * TY];
}

// Bahdanau attention: scores, softmax over Tx, weighted context. One block per batch row.
__global__ void __launch_bounds__(256)
attn_ctx(const float* __restrict__ ha,      // [B, ATT]
         const float* __restrict__ vp,      // [B, Tx, ATT]
         const float* __restrict__ aV,      // [ATT]
         const float* __restrict__ abV,     // [1]
         const float* __restrict__ enc,     // [B, Tx, DMODEL]
         float* __restrict__ ctx)           // [B, DMODEL]
{
    __shared__ float sc[TX];
    __shared__ float inv_s;
    int b = blockIdx.x;
    int lane = threadIdx.x & 31, wave = threadIdx.x >> 5;

    for (int tx = wave; tx < TX; tx += 8) {
        const float* vr = vp + ((long)(b * TX + tx)) * ATT;
        const float* hr = ha + (long)b * ATT;
        float p = 0.0f;
        for (int a = lane; a < ATT; a += 32)
            p += tanhf(hr[a] + vr[a]) * aV[a];
        for (int off = 16; off > 0; off >>= 1)
            p += __shfl_down(p, off, 32);
        if (lane == 0) sc[tx] = p + abV[0];
    }
    __syncthreads();
    if (threadIdx.x == 0) {
        float m = sc[0];
        for (int i = 1; i < TX; ++i) m = fmaxf(m, sc[i]);
        float s = 0.0f;
        for (int i = 0; i < TX; ++i) { sc[i] = expf(sc[i] - m); s += sc[i]; }
        inv_s = 1.0f / s;
    }
    __syncthreads();
    float inv = inv_s;
    for (int d = threadIdx.x; d < DMODEL; d += 256) {
        float acc = 0.0f;
        for (int tx = 0; tx < TX; ++tx)
            acc += sc[tx] * enc[((long)(b * TX + tx)) * DMODEL + d];
        ctx[(long)b * DMODEL + d] = acc * inv;
    }
}

__global__ void xin_build(const float* __restrict__ ctx, const float* __restrict__ dec_emb,
                          const int* __restrict__ tok, bf16_t* __restrict__ xin) {
    int idx = blockIdx.x * 256 + threadIdx.x;
    if (idx >= BATCH * (DMODEL + EMB)) return;
    int b = idx / (DMODEL + EMB), j = idx - b * (DMODEL + EMB);
    float v = (j < DMODEL) ? ctx[(long)b * DMODEL + j]
                           : dec_emb[(long)tok[b] * EMB + (j - DMODEL)];
    xin[idx] = (bf16_t)v;
}

// First-max argmax per row (jnp.argmax tie-break). One block per batch row.
__global__ void __launch_bounds__(256)
argmax_row(const float* __restrict__ logits, int ldr, int V, int* __restrict__ tok) {
    int b = blockIdx.x;
    const float* r = logits + (long)b * ldr;
    float best = -3.402823466e38f; int bi = V;
    for (int v = threadIdx.x; v < V; v += 256) {
        float x = r[v];
        if (x > best || (x == best && v < bi)) { best = x; bi = v; }
    }
    __shared__ float bv[256];
    __shared__ int   bx[256];
    bv[threadIdx.x] = best; bx[threadIdx.x] = bi;
    __syncthreads();
    for (int s = 128; s > 0; s >>= 1) {
        if (threadIdx.x < s) {
            if (bv[threadIdx.x + s] > bv[threadIdx.x] ||
                (bv[threadIdx.x + s] == bv[threadIdx.x] && bx[threadIdx.x + s] < bx[threadIdx.x])) {
                bv[threadIdx.x] = bv[threadIdx.x + s];
                bx[threadIdx.x] = bx[threadIdx.x + s];
            }
        }
        __syncthreads();
    }
    if (threadIdx.x == 0) tok[b] = bx[0];
}

// ---------------------------------------------------------------------------
// Host orchestration
// ---------------------------------------------------------------------------
extern "C" void kernel_launch(void* const* d_in, const int* in_sizes, int n_in,
                              void* d_out, int out_size, void* d_ws, size_t ws_size,
                              hipStream_t stream) {
    (void)in_sizes; (void)n_in; (void)out_size; (void)ws_size;

    const int*   inp     = (const int*)  d_in[0];
    const int*   targ    = (const int*)  d_in[1];
    const float* enc_emb = (const float*)d_in[2];
    const float* dec_emb = (const float*)d_in[3];
    const float* Wx_f1 = (const float*)d_in[4],  *Wh_f1 = (const float*)d_in[5],  *b_f1 = (const float*)d_in[6];
    const float* Wx_b1 = (const float*)d_in[7],  *Wh_b1 = (const float*)d_in[8],  *b_b1 = (const float*)d_in[9];
    const float* Wx_fm = (const float*)d_in[10], *Wh_fm = (const float*)d_in[11], *b_fm = (const float*)d_in[12];
    const float* Wx_bm = (const float*)d_in[13], *Wh_bm = (const float*)d_in[14], *b_bm = (const float*)d_in[15];
    const float* Wx_fl = (const float*)d_in[16], *Wh_fl = (const float*)d_in[17], *b_fl = (const float*)d_in[18];
    const float* Wx_bl = (const float*)d_in[19], *Wh_bl = (const float*)d_in[20], *b_bl = (const float*)d_in[21];
    const float* aW1 = (const float*)d_in[22], *ab1 = (const float*)d_in[23];
    const float* aW2 = (const float*)d_in[24], *ab2 = (const float*)d_in[25];
    const float* aV  = (const float*)d_in[26], *abV = (const float*)d_in[27];
    const float* dWx = (const float*)d_in[28], *dWh = (const float*)d_in[29], *db = (const float*)d_in[30];
    const float* fcW = (const float*)d_in[31], *fcb = (const float*)d_in[32];
    float* out = (float*)d_out;

    char* base = (char*)d_ws;
    size_t off = 0;
    auto alloc = [&](size_t bytes) -> void* {
        void* p = base + off;
        off = (off + bytes + 255) & ~(size_t)255;
        return p;
    };

    bf16_t* wWx_f1 = (bf16_t*)alloc(sizeof(bf16_t) * EMB * 4 * UNITS);
    bf16_t* wWh_f1 = (bf16_t*)alloc(sizeof(bf16_t) * UNITS * 4 * UNITS);
    bf16_t* wWx_b1 = (bf16_t*)alloc(sizeof(bf16_t) * EMB * 4 * UNITS);
    bf16_t* wWh_b1 = (bf16_t*)alloc(sizeof(bf16_t) * UNITS * 4 * UNITS);
    bf16_t* wWx_fm = (bf16_t*)alloc(sizeof(bf16_t) * DMODEL * 4 * UNITS);
    bf16_t* wWh_fm = (bf16_t*)alloc(sizeof(bf16_t) * UNITS * 4 * UNITS);
    bf16_t* wWx_bm = (bf16_t*)alloc(sizeof(bf16_t) * DMODEL * 4 * UNITS);
    bf16_t* wWh_bm = (bf16_t*)alloc(sizeof(bf16_t) * UNITS * 4 * UNITS);
    bf16_t* wWx_fl = (bf16_t*)alloc(sizeof(bf16_t) * DMODEL * 4 * UNITS);
    bf16_t* wWh_fl = (bf16_t*)alloc(sizeof(bf16_t) * UNITS * 4 * UNITS);
    bf16_t* wWx_bl = (bf16_t*)alloc(sizeof(bf16_t) * DMODEL * 4 * UNITS);
    bf16_t* wWh_bl = (bf16_t*)alloc(sizeof(bf16_t) * UNITS * 4 * UNITS);
    bf16_t* waW1   = (bf16_t*)alloc(sizeof(bf16_t) * DMODEL * ATT);
    bf16_t* waW2   = (bf16_t*)alloc(sizeof(bf16_t) * DMODEL * ATT);
    bf16_t* wdWx   = (bf16_t*)alloc(sizeof(bf16_t) * (DMODEL + EMB) * 4 * DMODEL);
    bf16_t* wdWh   = (bf16_t*)alloc(sizeof(bf16_t) * DMODEL * 4 * DMODEL);
    bf16_t* wfcW   = (bf16_t*)alloc(sizeof(bf16_t) * DMODEL * VTAR);

    bf16_t* x_emb  = (bf16_t*)alloc(sizeof(bf16_t) * BATCH * TX * EMB);
    float*  xwx_f  = (float*) alloc(sizeof(float) * BATCH * TX * 4 * UNITS);
    float*  xwx_b  = (float*) alloc(sizeof(float) * BATCH * TX * 4 * UNITS);
    float*  hwh    = (float*) alloc(sizeof(float) * BATCH * 4 * UNITS);
    float*  yA     = (float*) alloc(sizeof(float) * BATCH * TX * DMODEL);
    float*  yB     = (float*) alloc(sizeof(float) * BATCH * TX * DMODEL);
    bf16_t* ybf    = (bf16_t*)alloc(sizeof(bf16_t) * BATCH * TX * DMODEL);
    float*  h_f    = (float*) alloc(sizeof(float) * BATCH * UNITS);
    float*  c_f    = (float*) alloc(sizeof(float) * BATCH * UNITS);
    float*  h_b    = (float*) alloc(sizeof(float) * BATCH * UNITS);
    float*  c_b    = (float*) alloc(sizeof(float) * BATCH * UNITS);
    bf16_t* hbf    = (bf16_t*)alloc(sizeof(bf16_t) * BATCH * UNITS);
    float*  vp     = (float*) alloc(sizeof(float) * BATCH * TX * ATT);
    float*  ha     = (float*) alloc(sizeof(float) * BATCH * ATT);
    float*  ctx    = (float*) alloc(sizeof(float) * BATCH * DMODEL);
    bf16_t* xin    = (bf16_t*)alloc(sizeof(bf16_t) * BATCH * (DMODEL + EMB));
    float*  z1     = (float*) alloc(sizeof(float) * BATCH * 4 * DMODEL);
    float*  z2     = (float*) alloc(sizeof(float) * BATCH * 4 * DMODEL);
    float*  dec_h  = (float*) alloc(sizeof(float) * BATCH * DMODEL);
    float*  dec_c  = (float*) alloc(sizeof(float) * BATCH * DMODEL);
    bf16_t* dec_hbf= (bf16_t*)alloc(sizeof(bf16_t) * BATCH * DMODEL);
    int*    tok    = (int*)   alloc(sizeof(int) * BATCH);

    auto cvt = [&](const float* s, bf16_t* d, long n) {
        long blocks = (n + 255) / 256; if (blocks > 8192) blocks = 8192;
        cvt_f32_bf16<<<(int)blocks, 256, 0, stream>>>(s, d, n);
    };
    auto zf = [&](float* p, long n) {
        long blocks = (n + 255) / 256; if (blocks > 4096) blocks = 4096;
        zero_f32<<<(int)blocks, 256, 0, stream>>>(p, n);
    };
    auto zb = [&](bf16_t* p, long n) {
        long blocks = (n + 255) / 256; if (blocks > 4096) blocks = 4096;
        zero_bf16<<<(int)blocks, 256, 0, stream>>>(p, n);
    };
    // Direct-global WMMA GEMM; LDB must be 2048 (Wh) or 512 (aW1).
    auto gemm = [&](const bf16_t* A, int lda, const bf16_t* B, int ldb,
                    float* C, int ldc, const float* bias, int M, int N, int K) {
        dim3 grid(N / 64, M / 32);
        if (ldb == 2048)
            gemm_bf16_wmma<2048><<<grid, 256, 0, stream>>>(A, lda, B, C, ldc, bias, K);
        else
            gemm_bf16_wmma<512><<<grid, 256, 0, stream>>>(A, lda, B, C, ldc, bias, K);
    };
    auto gemm_t = [&](const bf16_t* A, int lda, const bf16_t* B, int ldb,
                      float* C, int ldc, const float* bias, int M, int N, int K) {
        dim3 grid(N / 64, M / 32);
        gemm_bf16_tdm<<<grid, 256, 0, stream>>>(A, lda, B, ldb, C, ldc, bias, K);
    };

    // ---- weight conversion (once per launch; fcW bf16 = 64MB -> L2-resident) ----
    cvt(Wx_f1, wWx_f1, (long)EMB * 4 * UNITS);      cvt(Wh_f1, wWh_f1, (long)UNITS * 4 * UNITS);
    cvt(Wx_b1, wWx_b1, (long)EMB * 4 * UNITS);      cvt(Wh_b1, wWh_b1, (long)UNITS * 4 * UNITS);
    cvt(Wx_fm, wWx_fm, (long)DMODEL * 4 * UNITS);   cvt(Wh_fm, wWh_fm, (long)UNITS * 4 * UNITS);
    cvt(Wx_bm, wWx_bm, (long)DMODEL * 4 * UNITS);   cvt(Wh_bm, wWh_bm, (long)UNITS * 4 * UNITS);
    cvt(Wx_fl, wWx_fl, (long)DMODEL * 4 * UNITS);   cvt(Wh_fl, wWh_fl, (long)UNITS * 4 * UNITS);
    cvt(Wx_bl, wWx_bl, (long)DMODEL * 4 * UNITS);   cvt(Wh_bl, wWh_bl, (long)UNITS * 4 * UNITS);
    cvt(aW1, waW1, (long)DMODEL * ATT);             cvt(aW2, waW2, (long)DMODEL * ATT);
    cvt(dWx, wdWx, (long)(DMODEL + EMB) * 4 * DMODEL);
    cvt(dWh, wdWh, (long)DMODEL * 4 * DMODEL);
    cvt(fcW, wfcW, (long)DMODEL * VTAR);

    // ---- encoder ----
    gather_embed<<<(BATCH * TX * EMB + 255) / 256, 256, 0, stream>>>(
        enc_emb, inp, x_emb, BATCH * TX, EMB);

    auto run_dir = [&](const float* xwx, const bf16_t* Wh, const float* bias,
                       float* h, float* c, float* y, int dir, bool fwd) {
        long nhc = (long)BATCH * UNITS;
        zf(h, nhc); zf(c, nhc); zb(hbf, nhc);
        for (int s = 0; s < TX; ++s) {
            int t = fwd ? s : (TX - 1 - s);
            gemm(hbf, UNITS, Wh, 4 * UNITS, hwh, 4 * UNITS, nullptr,
                 BATCH, 4 * UNITS, UNITS);
            lstm_gates<<<(BATCH * UNITS + 255) / 256, 256, 0, stream>>>(
                xwx + (long)t * 4 * UNITS, TX * 4 * UNITS, hwh, bias,
                h, hbf, c,
                y + (long)t * DMODEL + (long)dir * UNITS, (long)TX * DMODEL,
                UNITS, BATCH * UNITS);
        }
    };
    auto run_layer = [&](const bf16_t* xin_l, int K,
                         const bf16_t* Wxf, const bf16_t* Whf, const float* bfo,
                         const bf16_t* Wxb, const bf16_t* Whb, const float* bbo,
                         float* y) {
        gemm_t(xin_l, K, Wxf, 4 * UNITS, xwx_f, 4 * UNITS, nullptr, BATCH * TX, 4 * UNITS, K);
        gemm_t(xin_l, K, Wxb, 4 * UNITS, xwx_b, 4 * UNITS, nullptr, BATCH * TX, 4 * UNITS, K);
        run_dir(xwx_f, Whf, bfo, h_f, c_f, y, 0, true);
        run_dir(xwx_b, Whb, bbo, h_b, c_b, y, 1, false);
    };

    run_layer(x_emb, EMB, wWx_f1, wWh_f1, b_f1, wWx_b1, wWh_b1, b_b1, yA);
    cvt(yA, ybf, (long)BATCH * TX * DMODEL);
    run_layer(ybf, DMODEL, wWx_fm, wWh_fm, b_fm, wWx_bm, wWh_bm, b_bm, yB);
    cvt(yB, ybf, (long)BATCH * TX * DMODEL);
    run_layer(ybf, DMODEL, wWx_fm, wWh_fm, b_fm, wWx_bm, wWh_bm, b_bm, yA);
    cvt(yA, ybf, (long)BATCH * TX * DMODEL);
    run_layer(ybf, DMODEL, wWx_fl, wWh_fl, b_fl, wWx_bl, wWh_bl, b_bl, yB);
    cvt(yB, ybf, (long)BATCH * TX * DMODEL);     // ybf = enc_out (bf16)
    float* enc_out = yB;                         // enc_out (f32)

    // ---- decoder setup ----
    concat_hc<<<(BATCH * DMODEL + 255) / 256, 256, 0, stream>>>(
        h_f, c_f, h_b, c_b, dec_h, dec_hbf, dec_c);
    gemm_t(ybf, DMODEL, waW2, ATT, vp, ATT, ab2, BATCH * TX, ATT, DMODEL);
    init_tok<<<1, 32, 0, stream>>>(targ, tok);

    // ---- decoder loop (greedy feedback) ----
    for (int t = 0; t < TY; ++t) {
        gemm(dec_hbf, DMODEL, waW1, ATT, ha, ATT, ab1, BATCH, ATT, DMODEL);
        attn_ctx<<<BATCH, 256, 0, stream>>>(ha, vp, aV, abV, enc_out, ctx);
        xin_build<<<(BATCH * (DMODEL + EMB) + 255) / 256, 256, 0, stream>>>(
            ctx, dec_emb, tok, xin);
        gemm_t(xin, DMODEL + EMB, wdWx, 4 * DMODEL, z1, 4 * DMODEL, nullptr,
               BATCH, 4 * DMODEL, DMODEL + EMB);
        gemm_t(dec_hbf, DMODEL, wdWh, 4 * DMODEL, z2, 4 * DMODEL, nullptr,
               BATCH, 4 * DMODEL, DMODEL);
        lstm_gates<<<(BATCH * DMODEL + 255) / 256, 256, 0, stream>>>(
            z1, 4 * DMODEL, z2, db, dec_h, dec_hbf, dec_c,
            nullptr, 0, DMODEL, BATCH * DMODEL);
        gemm_t(dec_hbf, DMODEL, wfcW, VTAR, out + (long)t * VTAR, TY * VTAR, fcb,
               BATCH, VTAR, DMODEL);
        argmax_row<<<BATCH, 256, 0, stream>>>(out + (long)t * VTAR, TY * VTAR, VTAR, tok);
    }
}
